// MemoryConditionedAttention_2293512536675
// MI455X (gfx1250) — matrix operand order repaired and checked by reading, not compiled, not run on hardware
//
#include <hip/hip_runtime.h>
#include <stdint.h>

// Problem constants from the reference
#define NN 8192
#define DDIM 512
// H=4, QK=128 are irrelevant: softmax rows sum to 1 exactly, so
// attn.mean(heads).mean(keys) == 1/N -> Wq/Wk/QK^T/softmax are dead code.
// b_q = (sigmoid(r)/N) / (sum(sigmoid(r))/N + 1e-8); attended = diag(b_q)*(P @ Wout^T).

typedef __attribute__((ext_vector_type(16))) __bf16 v16bf;
typedef __attribute__((ext_vector_type(8)))  __bf16 v8bf;   // 16 bytes = 1 b128
typedef __attribute__((ext_vector_type(8)))  float  v8f;

union bffrag { v16bf v; v8bf h[2]; };

// ---------------- kernel 1: S = sum(sigmoid(residuals)) ----------------
__global__ __launch_bounds__(1024)
void sigsum_kernel(const float* __restrict__ res, float* __restrict__ ws) {
  __shared__ float smem[1024];
  int tid = threadIdx.x;
  float acc = 0.0f;
  for (int i = tid; i < NN; i += 1024) {
    float r = res[i];
    acc += 1.0f / (1.0f + __expf(-r));
  }
  smem[tid] = acc;
  __syncthreads();
  for (int s = 512; s > 0; s >>= 1) {
    if (tid < s) smem[tid] += smem[tid + s];
    __syncthreads();
  }
  if (tid == 0) ws[0] = smem[0];   // plain store: no ws init needed
}

// ---------------- kernel 2: b_q = (sig/N) / (S/N + 1e-8) ----------------
__global__ __launch_bounds__(256)
void bq_kernel(const float* __restrict__ res, const float* __restrict__ ws,
               float* __restrict__ bq) {
  int i = blockIdx.x * blockDim.x + threadIdx.x;
  if (i < NN) {
    float S   = ws[0];
    float sig = 1.0f / (1.0f + __expf(-res[i]));
    float num = sig * (1.0f / (float)NN);
    float den = S * (1.0f / (float)NN) + 1e-8f;
    bq[i] = num / den;
  }
}

// ---------------- kernel 3: fp32 -> bf16 staging (streaming, 8 elems/thread) ----
__global__ __launch_bounds__(256)
void cvt_bf16_kernel(const float* __restrict__ src, v8bf* __restrict__ dst, int n8) {
  int i = blockIdx.x * 256 + threadIdx.x;
  if (i >= n8) return;
  const float4* s = reinterpret_cast<const float4*>(src) + 2 * (size_t)i;
  float4 x = s[0], y = s[1];
  v8bf o;
  o[0] = (__bf16)x.x; o[1] = (__bf16)x.y; o[2] = (__bf16)x.z; o[3] = (__bf16)x.w;
  o[4] = (__bf16)y.x; o[5] = (__bf16)y.y; o[6] = (__bf16)y.z; o[7] = (__bf16)y.w;
  dst[i] = o;
}

// ---------------- kernel 4: attended = diag(b_q) * (Pb @ Wb^T), bf16 inputs ----
// Tiling: block = 8 waves, ALL waves share the same 64 output columns
// (B lines identical across waves -> L1 broadcast); wave w owns rows
// m0 = bx*128 + w*16. Inner loop: 10x global_load_b128 (bf16, zero converts)
// + 4x v_wmma_f32_16x16x32_bf16 per 32-wide K step.
__global__ __launch_bounds__(256)
void gemm_bf16_kernel(const __bf16* __restrict__ Pb,  // (NN, DDIM) bf16 row-major
                      const __bf16* __restrict__ Wb,  // (DDIM, DDIM) bf16 row-major
                      const float* __restrict__ bq,   // (NN,)
                      float* __restrict__ out) {      // (NN, DDIM) fp32
  const int wave = threadIdx.x >> 5;
  const int lane = threadIdx.x & 31;
  const int g    = lane >> 4;                   // lane half (0/1)
  const int h    = lane & 15;                   // index within half
  const int m0   = blockIdx.x * 128 + wave * 16;
  const int n0   = blockIdx.y * 64;

  // A: lane(g,h) -> row h; elems 0..15 = K=k0+16g..+15 (contig)
  const v8bf* pa = reinterpret_cast<const v8bf*>(
      Pb + (size_t)(m0 + h) * DDIM + 16 * g);
  // B[k][d] = Wout[d][k]: lane(g,h) -> col h; elems 0..15 = K=k0+16g..+15 (contig)
  const v8bf* pb0 = reinterpret_cast<const v8bf*>(
      Wb + (size_t)(n0 +  0 + h) * DDIM + 16 * g);
  const v8bf* pb1 = reinterpret_cast<const v8bf*>(
      Wb + (size_t)(n0 + 16 + h) * DDIM + 16 * g);
  const v8bf* pb2 = reinterpret_cast<const v8bf*>(
      Wb + (size_t)(n0 + 32 + h) * DDIM + 16 * g);
  const v8bf* pb3 = reinterpret_cast<const v8bf*>(
      Wb + (size_t)(n0 + 48 + h) * DDIM + 16 * g);

  v8f c0 = {}, c1 = {}, c2 = {}, c3 = {};

  #pragma unroll 2
  for (int k0 = 0; k0 < DDIM; k0 += 32) {
    const int kv = k0 >> 3;                      // offset in v8bf units
    bffrag a, b0, b1, b2, b3;
    a.h[0]  = pa[kv];     a.h[1]  = pa[kv + 1];  // +8 values
    b0.h[0] = pb0[kv];    b0.h[1] = pb0[kv + 1];
    b1.h[0] = pb1[kv];    b1.h[1] = pb1[kv + 1];
    b2.h[0] = pb2[kv];    b2.h[1] = pb2[kv + 1];
    b3.h[0] = pb3[kv];    b3.h[1] = pb3[kv + 1];

    c0 = __builtin_amdgcn_wmma_f32_16x16x32_bf16(false, a.v, false, b0.v,
                                                 (short)0, c0, false, false);
    c1 = __builtin_amdgcn_wmma_f32_16x16x32_bf16(false, a.v, false, b1.v,
                                                 (short)0, c1, false, false);
    c2 = __builtin_amdgcn_wmma_f32_16x16x32_bf16(false, a.v, false, b2.v,
                                                 (short)0, c2, false, false);
    c3 = __builtin_amdgcn_wmma_f32_16x16x32_bf16(false, a.v, false, b3.v,
                                                 (short)0, c3, false, false);
  }

  // Epilogue: C/D layout -> VGPR r of lane(g,h) holds D[m0+r+8g][n_tile+h].
  float scale[8];
  #pragma unroll
  for (int r = 0; r < 8; ++r) scale[r] = bq[m0 + r + 8 * g];

  #pragma unroll
  for (int r = 0; r < 8; ++r) {
    const size_t row = (size_t)(m0 + r + 8 * g);
    out[row * DDIM + n0 +  0 + h] = c0[r] * scale[r];
    out[row * DDIM + n0 + 16 + h] = c1[r] * scale[r];
    out[row * DDIM + n0 + 32 + h] = c2[r] * scale[r];
    out[row * DDIM + n0 + 48 + h] = c3[r] * scale[r];
  }
}

// ---------------- fallback GEMM (fp32 sources, convert in-loop) ----------------
// Used only if ws_size can't hold the bf16 staging buffers.
__device__ __forceinline__ void cvt8_to(v16bf& v, int base,
                                        const float4 x, const float4 y) {
  v[base + 0] = (__bf16)x.x; v[base + 1] = (__bf16)x.y;
  v[base + 2] = (__bf16)x.z; v[base + 3] = (__bf16)x.w;
  v[base + 4] = (__bf16)y.x; v[base + 5] = (__bf16)y.y;
  v[base + 6] = (__bf16)y.z; v[base + 7] = (__bf16)y.w;
}

__global__ __launch_bounds__(256)
void gemm_f32src_kernel(const float* __restrict__ P, const float* __restrict__ Wout,
                        const float* __restrict__ bq, float* __restrict__ out) {
  const int wave = threadIdx.x >> 5;
  const int lane = threadIdx.x & 31;
  const int g = lane >> 4, h = lane & 15;
  const int m0 = blockIdx.x * 128 + wave * 16;
  const int n0 = blockIdx.y * 64;

  const float* pa_base = P + (size_t)(m0 + h) * DDIM + 16 * g;
  const float* pb_base[4];
  #pragma unroll
  for (int t = 0; t < 4; ++t)
    pb_base[t] = Wout + (size_t)(n0 + t * 16 + h) * DDIM + 16 * g;

  v8f c[4] = {v8f{}, v8f{}, v8f{}, v8f{}};
  for (int k0 = 0; k0 < DDIM; k0 += 32) {
    const float4* pa  = reinterpret_cast<const float4*>(pa_base + k0);
    const float4* pa2 = reinterpret_cast<const float4*>(pa_base + k0 + 8);
    v16bf a;
    cvt8_to(a, 0, pa[0], pa[1]);
    cvt8_to(a, 8, pa2[0], pa2[1]);
    #pragma unroll
    for (int t = 0; t < 4; ++t) {
      const float4* pb = reinterpret_cast<const float4*>(pb_base[t] + k0);
      v16bf b;
      cvt8_to(b, 0, pb[0], pb[1]);
      cvt8_to(b, 8, pb[2], pb[3]);
      c[t] = __builtin_amdgcn_wmma_f32_16x16x32_bf16(false, a, false, b,
                                                     (short)0, c[t], false, false);
    }
  }
  float scale[8];
  #pragma unroll
  for (int r = 0; r < 8; ++r) scale[r] = bq[m0 + r + 8 * g];
  #pragma unroll
  for (int r = 0; r < 8; ++r) {
    const size_t row = (size_t)(m0 + r + 8 * g);
    #pragma unroll
    for (int t = 0; t < 4; ++t)
      out[row * DDIM + n0 + t * 16 + h] = c[t][r] * scale[r];
  }
}

// ---------------- launcher ----------------
extern "C" void kernel_launch(void* const* d_in, const int* in_sizes, int n_in,
                              void* d_out, int out_size, void* d_ws, size_t ws_size,
                              hipStream_t stream) {
  (void)in_sizes; (void)n_in; (void)out_size;
  const float* P    = (const float*)d_in[0];   // patch_features (N, D)
  const float* res  = (const float*)d_in[1];   // coreset_residuals (N,)
  // d_in[2] = Wq, d_in[3] = Wk: mathematically dead (softmax rows sum to 1)
  const float* Wout = (const float*)d_in[4];   // (D, D)

  float* out = (float*)d_out;                  // attended (N*D) then b_q (N)
  float* bq  = out + (size_t)NN * DDIM;
  float* S   = (float*)d_ws;                   // ws[0..3]: reduction scalar (16B slot)

  sigsum_kernel<<<1, 1024, 0, stream>>>(res, S);
  bq_kernel<<<NN / 256, 256, 0, stream>>>(res, S, bq);

  const size_t nP = (size_t)NN * DDIM, nW = (size_t)DDIM * DDIM;
  const size_t need = 16 + 2 * (nP + nW);      // bf16 staging after 16B scalar slot

  if (ws_size >= need) {
    __bf16* Pb = (__bf16*)((char*)d_ws + 16);
    __bf16* Wb = Pb + nP;
    cvt_bf16_kernel<<<(int)(nP / 8 / 256), 256, 0, stream>>>(P, (v8bf*)Pb, (int)(nP / 8));
    cvt_bf16_kernel<<<(int)(nW / 8 / 256), 256, 0, stream>>>(Wout, (v8bf*)Wb, (int)(nW / 8));
    gemm_bf16_kernel<<<dim3(NN / 128, DDIM / 64), 256, 0, stream>>>(Pb, Wb, bq, out);
  } else {
    gemm_f32src_kernel<<<dim3(NN / 128, DDIM / 64), 256, 0, stream>>>(P, Wout, bq, out);
  }
}
